// AdaptiveValuesMetadataAttention_5600637354235
// MI455X (gfx1250) — matrix operand order repaired and checked by reading, not compiled, not run on hardware
//
#include <hip/hip_runtime.h>
#include <hip/hip_bf16.h>

// Problem constants (B=1)
#define S_  8
#define N_  512
#define DV_ 256
#define DM_ 64
#define INNER_ 256
#define H_  8
#define DH_ 32
#define WS_ 3

typedef __attribute__((ext_vector_type(16))) _Float16 v16h;
typedef __attribute__((ext_vector_type(8)))  _Float16 v8h;
typedef __attribute__((ext_vector_type(8)))  float    v8f;

union V16H { v16h v; v8h h[2]; _Float16 e[16]; };
union V8F  { v8f  v; float e[8]; };
union V8H  { v8h  v; _Float16 e[8]; };

#define WMMA_F16(a, b, c) \
  __builtin_amdgcn_wmma_f32_16x16x32_f16(false, (a), false, (b), (short)0, (c), false, false)

// ---------- WMMA fragment loaders (wave32; A/B use contiguous k = 16*(lane/16)+s in both, not the ISA table order) ----------
// A: 16x32 (MxK): lane m = lane&15; lanes<16 hold K 0..15; lanes>=16 hold K 16..31.
// Each lane = two contiguous 8-half runs -> two b128 loads.
__device__ __forceinline__ v16h load_A_f16v(const _Float16* __restrict__ p, int ld) {
  int lane = threadIdx.x & 31;
  int m = lane & 15, hi = (lane >> 4) & 1;
  V16H a;
  a.h[0] = *(const v8h*)(p + (size_t)m * ld + (hi ? 16 : 0));
  a.h[1] = *(const v8h*)(p + (size_t)m * ld + (hi ? 24 : 8));
  return a.v;
}
// B: 32x16 (KxN) with element (k,n) at p[n*ld + k] (K contiguous in memory per n):
// lane n = lane&15; lanes<16: K 0..15, lanes>=16: K 16..31 -> two b128 loads.
__device__ __forceinline__ v16h load_Bk_f16v(const _Float16* __restrict__ p, int ld) {
  int lane = threadIdx.x & 31;
  int n = lane & 15, kb = (lane & 16) ? 16 : 0;
  const v8h* r = (const v8h*)(p + (size_t)n * ld + kb);
  V16H b;
  b.h[0] = r[0];
  b.h[1] = r[1];
  return b.v;
}
// C: 16x16 f32: VGPR r holds row m = r + (lane>=16 ? 8 : 0), col n = lane&15.
__device__ __forceinline__ void store_C_f16(_Float16* __restrict__ p, int ld, const V8F& c, bool clip) {
  int lane = threadIdx.x & 31;
  int n = lane & 15, mh = (lane & 16) ? 8 : 0;
#pragma unroll
  for (int r = 0; r < 8; ++r) {
    float v = c.e[r];
    if (clip) v = fminf(fmaxf(v, -5.0f), 5.0f);
    p[(size_t)(mh + r) * ld + n] = (_Float16)v;
  }
}
// Transposed C store: element (m,n) -> p[n*ld + m]. A lane's 8 rows are contiguous -> one b128 store.
__device__ __forceinline__ void store_CT_f16v(_Float16* __restrict__ p, int ld, const V8F& c) {
  int lane = threadIdx.x & 31;
  int n = lane & 15, mh = (lane & 16) ? 8 : 0;
  V8H pk;
#pragma unroll
  for (int r = 0; r < 8; ++r) pk.e[r] = (_Float16)c.e[r];
  *(v8h*)(p + (size_t)n * ld + mh) = pk.v;
}

// ---------- Kernel 0: pre-pass — f32->f16 convert activations, convert+transpose weights ----------
__global__ void prepass_kernel(const float* __restrict__ values,
                               const float* __restrict__ metadata,
                               const float* __restrict__ w_qkv,
                               const float* __restrict__ w_meta_inner,
                               const float* __restrict__ w_out,
                               _Float16* __restrict__ valh, _Float16* __restrict__ meth,
                               _Float16* __restrict__ wqkvT, _Float16* __restrict__ wmiT,
                               _Float16* __restrict__ woutT) {
  const int NV = S_ * N_ * DV_;        // 1048576
  const int NM = S_ * N_ * DM_;        // 262144
  const int NQ = DV_ * 3 * INNER_;     // 196608
  const int NI = DM_ * 2 * INNER_;     // 32768
  const int NO = INNER_ * DV_;         // 65536
  int i = blockIdx.x * 256 + threadIdx.x;
  if (i < NV) {
    valh[i] = (_Float16)values[i];
  } else if (i < NV + NM) {
    int j = i - NV;
    meth[j] = (_Float16)metadata[j];
  } else if (i < NV + NM + NQ) {
    int j = i - NV - NM;
    int k = j / (3 * INNER_), c = j % (3 * INNER_);
    wqkvT[(size_t)c * DV_ + k] = (_Float16)w_qkv[j];
  } else if (i < NV + NM + NQ + NI) {
    int j = i - NV - NM - NQ;
    int k = j / (2 * INNER_), c = j % (2 * INNER_);
    wmiT[(size_t)c * DM_ + k] = (_Float16)w_meta_inner[j];
  } else if (i < NV + NM + NQ + NI + NO) {
    int j = i - NV - NM - NQ - NI;
    int k = j / DV_, c = j % DV_;
    woutT[(size_t)c * INNER_ + k] = (_Float16)w_out[j];
  }
}

// ---------- Kernel 1: outer metadata attention + top-3 window selection ----------
__global__ void outer_topk_kernel(const float* __restrict__ metadata,
                                  const float* __restrict__ w_meta_outer,
                                  int* __restrict__ top_idx) {
  __shared__ float mm[S_][DM_];
  __shared__ float qm[S_][INNER_];
  __shared__ float km[S_][INNER_];
  __shared__ float dots[S_][S_];
  int tid = threadIdx.x;

  for (int p = tid; p < S_ * DM_; p += 256) {
    int s = p >> 6, d = p & 63;
    float acc = 0.0f;
    const float* base = metadata + (size_t)s * N_ * DM_ + d;
    for (int n = 0; n < N_; ++n) acc += base[(size_t)n * DM_];
    mm[s][d] = acc * (1.0f / (float)N_);
  }
  __syncthreads();

  for (int p = tid; p < S_ * INNER_; p += 256) {
    int s = p >> 8, j = p & 255;
    float aq = 0.0f, ak = 0.0f;
    for (int d = 0; d < DM_; ++d) {
      float m = mm[s][d];
      aq += m * w_meta_outer[d * 2 * INNER_ + j];
      ak += m * w_meta_outer[d * 2 * INNER_ + INNER_ + j];
    }
    qm[s][j] = fminf(fmaxf(aq, -5.0f), 5.0f);
    km[s][j] = fminf(fmaxf(ak, -5.0f), 5.0f);
  }
  __syncthreads();

  if (tid < S_ * S_) {
    int q = tid >> 3, k = tid & 7;
    float acc = 0.0f;
    for (int j = 0; j < INNER_; ++j) acc += qm[q][j] * km[k][j];
    dots[q][k] = acc * (1.0f / 16.0f);   // INNER^-0.5 = 1/16
  }
  __syncthreads();

  if (tid < S_) {
    int q = tid;
    float row[S_];
    float mx = -1e30f;
    for (int k = 0; k < S_; ++k) { row[k] = dots[q][k]; mx = fmaxf(mx, row[k]); }
    float sum = 0.0f;
    for (int k = 0; k < S_; ++k) { row[k] = __expf(row[k] - mx); sum += row[k]; }
    float inv = 1.0f / sum;
    for (int k = 0; k < S_; ++k) row[k] *= inv;
    row[q] += 2.0f;                      // + 2*eye
    for (int w = 0; w < WS_; ++w) {      // top-3, stable (first index wins ties)
      int best = 0; float bv = -1e30f;
      for (int k = 0; k < S_; ++k) if (row[k] > bv) { bv = row[k]; best = k; }
      top_idx[q * WS_ + w] = best;
      row[best] = -1e30f;
    }
  }
}

// ---------- Kernel 2: per-source projections (factored: 8 sources, not 24 windows) ----------
// Head-major outputs: qp/kp/qm/km layout [s][h][key][DH]; vvT layout [s][dh(256)][key(512)].
// Attention then reads fully contiguous 32-key x 64B panels per head.
__global__ void proj_kernel(const _Float16* __restrict__ valh,
                            const _Float16* __restrict__ meth,
                            const _Float16* __restrict__ wqkvT,
                            const _Float16* __restrict__ wmiT,
                            _Float16* __restrict__ qp, _Float16* __restrict__ kp,
                            _Float16* __restrict__ vvT, _Float16* __restrict__ qm,
                            _Float16* __restrict__ km) {
  const int wave = threadIdx.x >> 5;
  const size_t SB = (size_t)N_ * INNER_;     // per-source block (== H*N*DH)
  const size_t HB = (size_t)N_ * DH_;        // per-head block
  int t = blockIdx.x * 8 + wave;
  if (t < 12288) {                       // values @ w_qkv : 8 src * 32 mtiles * 48 ntiles
    int s = t / 1536, r = t % 1536, mt = r / 48, nt = r % 48;
    const _Float16* A = valh + (size_t)s * N_ * DV_ + (size_t)(mt * 16) * DV_;
    V8F acc = {};
#pragma unroll
    for (int k0 = 0; k0 < DV_; k0 += 32) {
      v16h a = load_A_f16v(A + k0, DV_);
      v16h b = load_Bk_f16v(wqkvT + (size_t)(nt * 16) * DV_ + k0, DV_);
      acc.v = WMMA_F16(a, b, acc.v);
    }
    if (nt < 32) {                       // qp / kp, head-major, clipped
      int ntq = (nt < 16) ? nt : (nt - 16);
      _Float16* dst = (nt < 16) ? qp : kp;
      int h = ntq >> 1, sub = (ntq & 1) * 16;
      store_C_f16(dst + (size_t)s * SB + (size_t)h * HB + (size_t)(mt * 16) * DH_ + sub,
                  DH_, acc, true);
    } else {                             // V stored transposed: vvT[s][dh][key]
      store_CT_f16v(vvT + (size_t)s * SB + (size_t)((nt - 32) * 16) * N_ + mt * 16, N_, acc);
    }
  } else {                               // metadata @ w_meta_inner : 8 * 32 * 32
    int t2 = t - 12288;
    int s = t2 / 1024, r = t2 % 1024, mt = r / 32, nt = r % 32;
    const _Float16* A = meth + (size_t)s * N_ * DM_ + (size_t)(mt * 16) * DM_;
    V8F acc = {};
#pragma unroll
    for (int k0 = 0; k0 < DM_; k0 += 32) {
      v16h a = load_A_f16v(A + k0, DM_);
      v16h b = load_Bk_f16v(wmiT + (size_t)(nt * 16) * DM_ + k0, DM_);
      acc.v = WMMA_F16(a, b, acc.v);
    }
    int ntq = (nt < 16) ? nt : (nt - 16);
    _Float16* dst = (nt < 16) ? qm : km;
    int h = ntq >> 1, sub = (ntq & 1) * 16;
    store_C_f16(dst + (size_t)s * SB + (size_t)h * HB + (size_t)(mt * 16) * DH_ + sub,
                DH_, acc, true);
  }
}

// ---------- Kernel 3: fused dual-score flash attention over gathered windows ----------
// block = 4 waves; wave owns a 16-query tile. DH=32 == WMMA K depth -> Q loaded once.
__global__ void attn_kernel(const int* __restrict__ top_idx,
                            const _Float16* __restrict__ qp, const _Float16* __restrict__ kp,
                            const _Float16* __restrict__ vvT, const _Float16* __restrict__ qm,
                            const _Float16* __restrict__ km, _Float16* __restrict__ attn_out) {
  __shared__ __align__(16) _Float16 Pt[4][16][32];  // per-wave P staging (C-frag -> A-frag)
  const int lane = threadIdx.x & 31;
  const int wave = threadIdx.x >> 5;
  int b = blockIdx.x;
  int s = b >> 6, h = (b >> 3) & 7, qc = b & 7;
  int q0 = qc * 64 + wave * 16;
  const size_t SB = (size_t)N_ * INNER_;
  const size_t HB = (size_t)N_ * DH_;

  int idx0 = top_idx[s * WS_ + 0];
  int idx1 = top_idx[s * WS_ + 1];
  int idx2 = top_idx[s * WS_ + 2];

  // queries come from the slot-0 window source; head-major -> contiguous 16x32 panel
  v16h qpA = load_A_f16v(qp + (size_t)idx0 * SB + (size_t)h * HB + (size_t)q0 * DH_, DH_);
  v16h qmA = load_A_f16v(qm + (size_t)idx0 * SB + (size_t)h * HB + (size_t)q0 * DH_, DH_);

  float rowmax[8], rowsum[8];
  V8F oa = {}, ob = {};
#pragma unroll
  for (int r = 0; r < 8; ++r) { rowmax[r] = -1e30f; rowsum[r] = 0.0f; }
  const float scale = 0.17677669529663687f;   // DH^-0.5

  for (int c = 0; c < 48; ++c) {              // 48 chunks of 32 keys = 1536
    int w = c >> 4;
    int src = (w == 0) ? idx0 : ((w == 1) ? idx1 : idx2);
    int kr0 = (c & 15) * 32;
    const _Float16* Kb = kp  + (size_t)src * SB + (size_t)h * HB + (size_t)kr0 * DH_;
    const _Float16* Mb = km  + (size_t)src * SB + (size_t)h * HB + (size_t)kr0 * DH_;
    const _Float16* Vb = vvT + (size_t)src * SB + (size_t)(h * DH_) * N_ + kr0;
    if (c < 47) {                             // pull next contiguous 2KB K-chunk toward L0
      __builtin_prefetch(Kb + (size_t)32 * DH_, 0, 1);
      __builtin_prefetch(Vb + 32, 0, 1);
    }

    // scores: (qp.kp^T + qm.km^T) for two 16-key N-tiles; K^T frag = contiguous per-lane b128 pair
    V8F c0 = {}, c1 = {};
    c0.v = WMMA_F16(qpA, load_Bk_f16v(Kb, DH_), c0.v);
    c0.v = WMMA_F16(qmA, load_Bk_f16v(Mb, DH_), c0.v);
    c1.v = WMMA_F16(qpA, load_Bk_f16v(Kb + (size_t)16 * DH_, DH_), c1.v);
    c1.v = WMMA_F16(qmA, load_Bk_f16v(Mb + (size_t)16 * DH_, DH_), c1.v);

    // online softmax update (rows live in 16-lane half-groups; xor masks stay in-group)
#pragma unroll
    for (int r = 0; r < 8; ++r) {
      float s0 = c0.e[r] * scale, s1 = c1.e[r] * scale;
      float mx = fmaxf(s0, s1);
      mx = fmaxf(mx, __shfl_xor(mx, 1));
      mx = fmaxf(mx, __shfl_xor(mx, 2));
      mx = fmaxf(mx, __shfl_xor(mx, 4));
      mx = fmaxf(mx, __shfl_xor(mx, 8));
      float nm  = fmaxf(rowmax[r], mx);
      float fac = __expf(rowmax[r] - nm);
      float p0  = __expf(s0 - nm);
      float p1  = __expf(s1 - nm);
      float ps  = p0 + p1;
      ps += __shfl_xor(ps, 1);
      ps += __shfl_xor(ps, 2);
      ps += __shfl_xor(ps, 4);
      ps += __shfl_xor(ps, 8);
      rowsum[r] = rowsum[r] * fac + ps;
      rowmax[r] = nm;
      oa.e[r] *= fac;
      ob.e[r] *= fac;
      int m = r + ((lane & 16) ? 8 : 0);
      int n = lane & 15;
      Pt[wave][m][n]      = (_Float16)p0;
      Pt[wave][m][16 + n] = (_Float16)p1;
    }
    // P (16x32) x V (32x16, two DH halves). V^T layout -> contiguous b128 frags.
    // Same-wave LDS store->load is DS-ordered, no barrier needed.
    v16h pA = load_A_f16v(&Pt[wave][0][0], 32);
    oa.v = WMMA_F16(pA, load_Bk_f16v(Vb, N_), oa.v);
    ob.v = WMMA_F16(pA, load_Bk_f16v(Vb + (size_t)16 * N_, N_), ob.v);
  }

  // head-major output: attn[s][h][key][DH]
  int n = lane & 15, mh = (lane & 16) ? 8 : 0;
#pragma unroll
  for (int r = 0; r < 8; ++r) {
    float inv = 1.0f / rowsum[r];
    size_t base = (size_t)s * SB + (size_t)h * HB + (size_t)(q0 + mh + r) * DH_;
    attn_out[base + n]      = (_Float16)(oa.e[r] * inv);
    attn_out[base + 16 + n] = (_Float16)(ob.e[r] * inv);
  }
}

// ---------- Kernel 4: output projection + bias (f32 out) ----------
// attn is head-major [s][h][key][DH]; each 32-wide K chunk sits inside one head block,
// so A fragments remain two contiguous b128 loads per lane (ld = DH).
__global__ void outproj_kernel(const _Float16* __restrict__ attn_in,
                               const _Float16* __restrict__ woutT,
                               const float* __restrict__ b_out,
                               float* __restrict__ out) {
  const int lane = threadIdx.x & 31;
  const int wave = threadIdx.x >> 5;
  const size_t SB = (size_t)N_ * INNER_;
  const size_t HB = (size_t)N_ * DH_;
  int t = blockIdx.x * 8 + wave;               // 8 s * 32 mt * 16 nt = 4096
  int s = t / 512, r = t % 512, mt = r / 16, nt = r % 16;
  V8F acc = {};
#pragma unroll
  for (int k0 = 0; k0 < INNER_; k0 += 32) {
    const _Float16* A = attn_in + (size_t)s * SB + (size_t)(k0 >> 5) * HB + (size_t)(mt * 16) * DH_;
    v16h a = load_A_f16v(A, DH_);
    v16h b = load_Bk_f16v(woutT + (size_t)(nt * 16) * INNER_ + k0, INNER_);
    acc.v = WMMA_F16(a, b, acc.v);
  }
  int n = lane & 15, mh = (lane & 16) ? 8 : 0;
  float bias = b_out[nt * 16 + n];
#pragma unroll
  for (int rr = 0; rr < 8; ++rr) {
    out[(size_t)(s * N_ + mt * 16 + mh + rr) * DV_ + nt * 16 + n] = acc.e[rr] + bias;
  }
}

extern "C" void kernel_launch(void* const* d_in, const int* in_sizes, int n_in,
                              void* d_out, int out_size, void* d_ws, size_t ws_size,
                              hipStream_t stream) {
  const float* values       = (const float*)d_in[0];
  const float* metadata     = (const float*)d_in[1];
  const float* w_meta_outer = (const float*)d_in[2];
  const float* w_qkv        = (const float*)d_in[3];
  const float* w_meta_inner = (const float*)d_in[4];
  const float* w_out        = (const float*)d_in[5];
  const float* b_out        = (const float*)d_in[6];
  float* out = (float*)d_out;

  // workspace layout (halfs unless noted):
  // [top_idx ints | qp | kp | qm | km | vvT | attn | valh | meth | wqkvT | wmiT | woutT]
  char* ws = (char*)d_ws;
  int* top_idx = (int*)ws;
  _Float16* hbase = (_Float16*)(ws + 256);
  const size_t MAT = (size_t)S_ * N_ * INNER_;   // 1,048,576 halfs
  _Float16* qp    = hbase;
  _Float16* kp    = qp + MAT;
  _Float16* qm    = kp + MAT;
  _Float16* km    = qm + MAT;
  _Float16* vvT   = km + MAT;
  _Float16* attn  = vvT + MAT;
  _Float16* valh  = attn + MAT;
  _Float16* meth  = valh + MAT;                           // S*N*DM = 262144
  _Float16* wqkvT = meth + (size_t)S_ * N_ * DM_;         // 196608
  _Float16* wmiT  = wqkvT + (size_t)DV_ * 3 * INNER_;     // 32768
  _Float16* woutT = wmiT + (size_t)DM_ * 2 * INNER_;      // 65536

  // total pre-pass elements: 1048576+262144+196608+32768+65536 = 1,605,632 = 6272*256
  hipLaunchKernelGGL(prepass_kernel,    dim3(6272), dim3(256), 0, stream,
                     values, metadata, w_qkv, w_meta_inner, w_out,
                     valh, meth, wqkvT, wmiT, woutT);
  hipLaunchKernelGGL(outer_topk_kernel, dim3(1),    dim3(256), 0, stream,
                     metadata, w_meta_outer, top_idx);
  hipLaunchKernelGGL(proj_kernel,       dim3(2560), dim3(256), 0, stream,
                     valh, meth, wqkvT, wmiT, qp, kp, vvT, qm, km);
  hipLaunchKernelGGL(attn_kernel,       dim3(512),  dim3(128), 0, stream,
                     top_idx, qp, kp, vvT, qm, km, attn);
  hipLaunchKernelGGL(outproj_kernel,    dim3(512),  dim3(256), 0, stream,
                     attn, woutT, b_out, out);
}
